// YatDense_16819091931202
// MI455X (gfx1250) — matrix operand order repaired and checked by reading, not compiled, not run on hardware
//
#include <hip/hip_runtime.h>
#include <math.h>

// ---------------------------------------------------------------------------
// YatDense for MI455X (gfx1250, wave32, WMMA), two-pass bf16x3 scheme.
//
// Pass 1: split f32 X and W into bf16 hi/lo arrays (hi = RNE bf16,
//         lo = bf16(f - hi)) and compute per-row squared norms. One-time
//         ~335 MB of traffic (~15 us at 23.3 TB/s).
// Pass 2: pure bf16 GEMM, y ~= hi*hi + hi*lo + lo*hi on
//         V_WMMA_F32_16X16X32_BF16 (3 WMMAs per K=32 vs 8 f32-WMMAs).
//         Fragments load straight from row-major bf16 via b128 loads:
//         CDNA5 A/B fragment layouts are contiguous-K runs per lane
//         (ISA 7.12.2). Per-lane BYTE offsets (32-bit) + uniform base
//         pointer -> saddr+voffset GVS addressing with immediate offsets,
//         so the k-loop has ~zero VALU. k-loop unrolled x2 for WMMA/load
//         overlap and to dodge the WMMA->VALU WAR hazard window.
//
// Fallback: if ws_size < ~151 MB, fused single-pass kernel.
// ---------------------------------------------------------------------------

typedef __attribute__((ext_vector_type(16))) __bf16 v16bf;
typedef __attribute__((ext_vector_type(8)))  __bf16 bf16x8;
typedef __attribute__((ext_vector_type(8)))  float  v8f;
typedef __attribute__((ext_vector_type(4)))  float  f32x4;

#define YAT_EPS 1e-6f

__device__ __forceinline__ v8f wmma_bf16(v16bf a, v16bf b, v8f c) {
  return __builtin_amdgcn_wmma_f32_16x16x32_bf16(false, a, false, b, (short)0, c,
                                                 false, false);
}

__device__ __forceinline__ bf16x8 ldb128(const char* base, unsigned voff) {
  return *(const bf16x8*)(base + voff);
}

// ======================= Pass 1: split + row norms =========================
__global__ __launch_bounds__(256) void split_bf16_rows(
    const float* __restrict__ src,  // [rows, K] f32
    __bf16* __restrict__ hi,        // [rows, K] bf16
    __bf16* __restrict__ lo,        // [rows, K] bf16
    float* __restrict__ sq,         // [rows]
    int rows, int K) {
  const int lane = threadIdx.x & 31;
  const int wave = threadIdx.x >> 5;
  const int row  = blockIdx.x * 8 + wave;
  if (row >= rows) return;

  const float* s = src + (size_t)row * (size_t)K;
  __bf16* ph = hi + (size_t)row * (size_t)K;
  __bf16* pl = lo + (size_t)row * (size_t)K;

  float acc = 0.0f;
  for (int k = lane * 8; k < K; k += 32 * 8) {
    union { f32x4 q[2]; float f[8]; } u;
    u.q[0] = ((const f32x4*)(s + k))[0];
    u.q[1] = ((const f32x4*)(s + k))[1];
    bf16x8 vh, vl;
#pragma unroll
    for (int i = 0; i < 8; ++i) {
      float f = u.f[i];
      acc += f * f;
      __bf16 h = (__bf16)f;
      vh[i] = h;
      vl[i] = (__bf16)(f - (float)h);
    }
    *(bf16x8*)(ph + k) = vh;
    *(bf16x8*)(pl + k) = vl;
  }
#pragma unroll
  for (int d = 16; d > 0; d >>= 1) acc += __shfl_xor(acc, d, 32);
  if (lane == 0) sq[row] = acc;
}

// ===================== Pass 2: bf16x3 GEMM + epilogue ======================
// Block = 8 waves (2 along M x 4 along N), wave tile 64x32, block 128x128.
__global__ __launch_bounds__(256) void yat_gemm_bf16x3(
    const __bf16* __restrict__ Xh, const __bf16* __restrict__ Xl, // [M,K]
    const __bf16* __restrict__ Wh, const __bf16* __restrict__ Wl, // [N,K]
    const float* __restrict__ xsq,   // [M]
    const float* __restrict__ wsq,   // [N]
    const float* __restrict__ alpha, // [1]
    const float* __restrict__ bias,  // [N]
    float* __restrict__ Out,         // [M,N]
    int K, int N) {
  const int lane = threadIdx.x & 31;
  const int wave = threadIdx.x >> 5;
  const int l16  = lane & 15;
  const int half = lane >> 4;

  const int wm = wave & 1;
  const int wn = wave >> 1;
  const int m0 = blockIdx.y * 128 + wm * 64;
  const int n0 = blockIdx.x * 128 + wn * 32;

  v8f acc[4][2];
#pragma unroll
  for (int i = 0; i < 4; ++i)
#pragma unroll
    for (int j = 0; j < 2; ++j) acc[i][j] = (v8f)(0.0f);

  // Per-lane 32-bit BYTE offsets (arrays <= 67 MB, fits easily). Uniform
  // char* bases stay in SGPRs -> saddr+voffset global_load_b128 form.
  const char* xh = (const char*)Xh;
  const char* xl = (const char*)Xl;
  const char* wh = (const char*)Wh;
  const char* wl = (const char*)Wl;

  unsigned aoffB[4], boffB[2];
#pragma unroll
  for (int i = 0; i < 4; ++i)
    aoffB[i] = ((unsigned)(m0 + 16 * i + l16) * (unsigned)K +
                (unsigned)(8 * half)) * 2u;
#pragma unroll
  for (int j = 0; j < 2; ++j)
    boffB[j] = ((unsigned)(n0 + 16 * j + l16) * (unsigned)K +
                (unsigned)(8 * half)) * 2u;

  const int KB = K * 2;  // row bytes
#pragma unroll 2
  for (int kb = 0; kb < KB; kb += 64) {
    v16bf Ah[4], Al[4], Bh[2], Bl[2];
#pragma unroll
    for (int i = 0; i < 4; ++i) {
      // A frag (16x32): lane K runs [k+8h, +8) and [k+16+8h, +8)  (x2 bytes)
      const unsigned o = aoffB[i] + (unsigned)kb;
      union { v16bf v; bf16x8 h[2]; } uh, ul;
      uh.h[0] = ldb128(xh, o);
      uh.h[1] = ldb128(xh, o + 32);
      ul.h[0] = ldb128(xl, o);
      ul.h[1] = ldb128(xl, o + 32);
      Ah[i] = uh.v; Al[i] = ul.v;
    }
#pragma unroll
    for (int j = 0; j < 2; ++j) {
      // B frag (32x16): lane K runs [k+8h, +8) and [k+16+8h, +8) of the W row (= B column)
      const unsigned o = boffB[j] + (unsigned)kb;
      union { v16bf v; bf16x8 h[2]; } uh, ul;
      uh.h[0] = ldb128(wh, o);
      uh.h[1] = ldb128(wh, o + 32);
      ul.h[0] = ldb128(wl, o);
      ul.h[1] = ldb128(wl, o + 32);
      Bh[j] = uh.v; Bl[j] = ul.v;
    }

#pragma unroll
    for (int i = 0; i < 4; ++i)
#pragma unroll
      for (int j = 0; j < 2; ++j) {
        acc[i][j] = wmma_bf16(Ah[i], Bh[j], acc[i][j]);
        acc[i][j] = wmma_bf16(Ah[i], Bl[j], acc[i][j]);
        acc[i][j] = wmma_bf16(Al[i], Bh[j], acc[i][j]);
      }
  }

  // ---- epilogue -----------------------------------------------------------
  float xsv[4];
#pragma unroll
  for (int i = 0; i < 4; ++i) xsv[i] = xsq[m0 + 16 * i + l16];
  float wsv[2];
#pragma unroll
  for (int j = 0; j < 2; ++j) wsv[j] = wsq[n0 + 16 * j + l16];

  const float fN    = (float)N;
  const float scale = powf(sqrtf(fN) / logf(1.0f + fN), alpha[0]);

#pragma unroll
  for (int j = 0; j < 2; ++j) {
    const int   n  = n0 + 16 * j + l16;
    const float ws = wsv[j];
    const float bn = bias[n];
#pragma unroll
    for (int i = 0; i < 4; ++i) {
      const int mbase = m0 + 16 * i + 8 * half;  // C/D: VGPR r -> row r+8*half
      float* orow = Out + (size_t)mbase * (size_t)N + n;
#pragma unroll
      for (int r = 0; r < 8; ++r) {
        float y  = acc[i][j][r];
        float xs = __shfl(xsv[i], r + 8 * half, 32);
        float d  = xs + ws - 2.0f * y + YAT_EPS;
        orow[(size_t)r * (size_t)N] = (y * y) / d * scale + bn;
      }
    }
  }
}

// ================== Fallback: fused single pass ============================
__device__ __forceinline__ float cvt16(const float* __restrict__ p0,
                                       const float* __restrict__ p1,
                                       v16bf& hi, v16bf& lo) {
  union { f32x4 q[4]; float f[16]; } u;
  u.q[0] = ((const f32x4*)p0)[0];
  u.q[1] = ((const f32x4*)p0)[1];
  u.q[2] = ((const f32x4*)p1)[0];
  u.q[3] = ((const f32x4*)p1)[1];
  float s = 0.0f;
#pragma unroll
  for (int i = 0; i < 16; ++i) {
    float f = u.f[i];
    s += f * f;
    __bf16 h = (__bf16)f;
    hi[i] = h;
    lo[i] = (__bf16)(f - (float)h);
  }
  return s;
}

__global__ __launch_bounds__(256) void yat_dense_fused(
    const float* __restrict__ X, const float* __restrict__ W,
    const float* __restrict__ alpha, const float* __restrict__ bias,
    float* __restrict__ Out, int M, int K, int N) {
  const int lane = threadIdx.x & 31;
  const int wave = threadIdx.x >> 5;
  const int l16  = lane & 15;
  const int half = lane >> 4;
  const int wm = wave & 3, wn = wave >> 2;
  const int m0 = blockIdx.y * 128 + wm * 32;
  const int n0 = blockIdx.x * 64  + wn * 32;

  v8f acc[2][2];
#pragma unroll
  for (int i = 0; i < 2; ++i)
#pragma unroll
    for (int j = 0; j < 2; ++j) acc[i][j] = (v8f)(0.0f);
  float xsqp[2] = {0.0f, 0.0f}, wsqp[2] = {0.0f, 0.0f};

  const float* xrow0 = X + (size_t)(m0 + l16) * (size_t)K;
  const float* xrow1 = xrow0 + (size_t)16 * (size_t)K;
  const float* wrow0 = W + (size_t)(n0 + l16) * (size_t)K;
  const float* wrow1 = wrow0 + (size_t)16 * (size_t)K;

  for (int k0 = 0; k0 < K; k0 += 32) {
    v16bf ahi[2], alo[2], bhi[2], blo[2];
    const float* pa0 = xrow0 + k0 + 8 * half;
    xsqp[0] += cvt16(pa0, pa0 + 16, ahi[0], alo[0]);
    const float* pa1 = xrow1 + k0 + 8 * half;
    xsqp[1] += cvt16(pa1, pa1 + 16, ahi[1], alo[1]);
    const float* pb0 = wrow0 + k0 + 8 * half;
    wsqp[0] += cvt16(pb0, pb0 + 16, bhi[0], blo[0]);
    const float* pb1 = wrow1 + k0 + 8 * half;
    wsqp[1] += cvt16(pb1, pb1 + 16, bhi[1], blo[1]);
#pragma unroll
    for (int i = 0; i < 2; ++i)
#pragma unroll
      for (int j = 0; j < 2; ++j) {
        acc[i][j] = wmma_bf16(ahi[i], bhi[j], acc[i][j]);
        acc[i][j] = wmma_bf16(ahi[i], blo[j], acc[i][j]);
        acc[i][j] = wmma_bf16(alo[i], bhi[j], acc[i][j]);
      }
  }

  float xst[2], wst[2];
#pragma unroll
  for (int i = 0; i < 2; ++i) {
    xst[i] = xsqp[i] + __shfl_xor(xsqp[i], 16, 32);
    wst[i] = wsqp[i] + __shfl_xor(wsqp[i], 16, 32);
  }
  const float fN    = (float)N;
  const float scale = powf(sqrtf(fN) / logf(1.0f + fN), alpha[0]);
#pragma unroll
  for (int j = 0; j < 2; ++j) {
    const int   n  = n0 + j * 16 + l16;
    const float ws = wst[j];
    const float bn = bias[n];
#pragma unroll
    for (int i = 0; i < 2; ++i) {
      const int mbase = m0 + i * 16 + 8 * half;
      float* orow = Out + (size_t)mbase * (size_t)N + n;
#pragma unroll
      for (int r = 0; r < 8; ++r) {
        float y  = acc[i][j][r];
        float xs = __shfl(xst[i], r + 8 * half, 32);
        float d  = xs + ws - 2.0f * y + YAT_EPS;
        orow[(size_t)r * (size_t)N] = (y * y) / d * scale + bn;
      }
    }
  }
}

// ============================== launcher ===================================
extern "C" void kernel_launch(void* const* d_in, const int* in_sizes, int n_in,
                              void* d_out, int out_size, void* d_ws, size_t ws_size,
                              hipStream_t stream) {
  (void)n_in; (void)out_size;
  const float* X     = (const float*)d_in[0];  // [M,K]
  const float* W     = (const float*)d_in[1];  // [N,K]
  const float* alpha = (const float*)d_in[2];  // [1]
  const float* bias  = (const float*)d_in[3];  // [N]
  float* out = (float*)d_out;

  const int F = in_sizes[3];            // N = 2048
  const int D = in_sizes[1] / F;        // K = 2048
  const int M = in_sizes[0] / D;        // M = 16384

  const size_t szX = (size_t)M * (size_t)D * 2;  // bf16 bytes
  const size_t szW = (size_t)F * (size_t)D * 2;
  const size_t need = 2 * szX + 2 * szW + (size_t)M * 4 + (size_t)F * 4;

  if (ws_size >= need) {
    char* p = (char*)d_ws;
    __bf16* Xh = (__bf16*)p;           p += szX;
    __bf16* Xl = (__bf16*)p;           p += szX;
    __bf16* Wh = (__bf16*)p;           p += szW;
    __bf16* Wl = (__bf16*)p;           p += szW;
    float*  xs = (float*)p;            p += (size_t)M * 4;
    float*  wsq = (float*)p;

    split_bf16_rows<<<dim3((M + 7) / 8), 256, 0, stream>>>(X, Xh, Xl, xs, M, D);
    split_bf16_rows<<<dim3((F + 7) / 8), 256, 0, stream>>>(W, Wh, Wl, wsq, F, D);
    yat_gemm_bf16x3<<<dim3(F / 128, M / 128), 256, 0, stream>>>(
        Xh, Xl, Wh, Wl, xs, wsq, alpha, bias, out, D, F);
  } else {
    yat_dense_fused<<<dim3(F / 64, M / 128), 256, 0, stream>>>(
        X, W, alpha, bias, out, M, D, F);
  }
}
